// minLSTMCell_7902739824853
// MI455X (gfx1250) — matrix operand order repaired and checked by reading, not compiled, not run on hardware
//
#include <hip/hip_runtime.h>
#include <hip/hip_bf16.h>

typedef __bf16 bf16;
typedef __attribute__((ext_vector_type(16))) __bf16 v16bf;
typedef __attribute__((ext_vector_type(8)))  __bf16 v8bf;
typedef __attribute__((ext_vector_type(8)))  float  v8f;
typedef __attribute__((ext_vector_type(4)))  int    v4i;

union Frag16 { v16bf v; v8bf h[2]; };

#define B_SZ 4
#define S_SZ 4096
#define D_SZ 1024
#define H_SZ 1024
#define M_SZ (B_SZ * S_SZ)     // 16384 rows
#define KDIM D_SZ
#define MB 256                 // block M tile
#define NB 64                  // block N tile
#define NCHUNK 32
#define CHLEN  (S_SZ / NCHUNK) // 128
#define EPSV 1e-8f

// ---------- gfx1250 async global->LDS staging (guarded, with sync fallback)
#if defined(__has_builtin)
#if __has_builtin(__builtin_amdgcn_global_load_async_to_lds_b128) && \
    __has_builtin(__builtin_amdgcn_s_wait_asynccnt)
#define ASYNC_STAGE 1
#endif
#endif
#ifndef ASYNC_STAGE
#define ASYNC_STAGE 0
#endif

#if ASYNC_STAGE
typedef __attribute__((address_space(1))) v4i* gptr_v4i;   // global
typedef __attribute__((address_space(3))) v4i* lptr_v4i;   // LDS
#endif

__device__ __forceinline__ void cp_b128(const bf16* g, bf16* l) {
#if ASYNC_STAGE
    __builtin_amdgcn_global_load_async_to_lds_b128(
        (gptr_v4i)const_cast<bf16*>(g), (lptr_v4i)l, 0, 0);
#else
    *(uint4*)l = *(const uint4*)g;
#endif
}

__device__ __forceinline__ void wait_async_stage() {
#if ASYNC_STAGE
    __builtin_amdgcn_s_wait_asynccnt(0);
#endif
}

// ---------- fp32 -> bf16 round-to-nearest-even, bit level
__device__ __forceinline__ unsigned short f2bfbits(float f) {
    union { float f; unsigned u; } v; v.f = f;
    unsigned r = v.u + 0x7FFFu + ((v.u >> 16) & 1u);
    return (unsigned short)(r >> 16);
}

__global__ void cvt_bf16_kernel(const float* __restrict__ src,
                                bf16* __restrict__ dst, int n4) {
    int i = blockIdx.x * blockDim.x + threadIdx.x;
    int stride = gridDim.x * blockDim.x;
    const float4* s4 = (const float4*)src;
    uint2* d2 = (uint2*)dst;
    for (; i < n4; i += stride) {
        float4 v = s4[i];
        unsigned p0 = (unsigned)f2bfbits(v.x) | ((unsigned)f2bfbits(v.y) << 16);
        unsigned p1 = (unsigned)f2bfbits(v.z) | ((unsigned)f2bfbits(v.w) << 16);
        d2[i] = make_uint2(p0, p1);
    }
}

// ---------- fused dual-GEMM (f and i gates) + in-register gate epilogue -----
// Block: 256 threads = 8 waves. Block tile: 256(M) x 64(N), K-step 32.
// Wave wv owns rows [wv*32, wv*32+32) and computes BOTH gemms for them:
// 16 accumulator tiles (2 gemm x 2 rowTile x 4 nTile), B-frag reused across
// row tiles, f/i pre-activations land in the same lanes -> register epilogue.
__global__ __launch_bounds__(256, 1)
void gemm_gates_kernel(const bf16* __restrict__ xb,
                       const bf16* __restrict__ wfb,
                       const bf16* __restrict__ wib,
                       const float* __restrict__ bias_f,
                       const float* __restrict__ bias_i,
                       float* __restrict__ a_buf,
                       float* __restrict__ b_out)
{
    __shared__ bf16 lds_x[2][MB][32];        // 32 KB (double buffered)
    __shared__ bf16 lds_w[2][2][NB][32];     // 16 KB

    const int tid   = threadIdx.x;
    const int lane  = tid & 31;
    const int wv    = tid >> 5;              // 0..7
    const int mBase = blockIdx.y * MB;
    const int nBase = blockIdx.x * NB;

    v8f acc[2][2][4] = {};                   // [gemm][rowTile][nTile]

    // A fragment addressing (16-bit A 16x32 layout, ISA 7.12.2)
    const int rowA = lane & 15;
    const int kbA  = (lane >> 4) * 8;        // lanes 0-15: K 0..7/16..23; 16-31: 8..15/24..31
    // B fragment addressing (B striped by N across lanes)
    const int bN   = lane & 15;
    const int kbB  = (lane >> 4) * 8;        // lanes 0-15: K 0..7/16..23; 16-31: 8..15/24..31

    auto stage = [&](int k0, int buf) {
        // x tile: 256x32 bf16 = 1024 16B segments, 4 per thread
        #pragma unroll
        for (int t = 0; t < 4; ++t) {
            int s = t * 256 + tid;
            int row = s >> 2, c8 = (s & 3) * 8;
            cp_b128(&xb[(size_t)(mBase + row) * KDIM + k0 + c8],
                    &lds_x[buf][row][c8]);
        }
        // both W tiles: 2x64x32 bf16 = 512 segments, 2 per thread
        #pragma unroll
        for (int t = 0; t < 2; ++t) {
            int s = t * 256 + tid;
            int g = s >> 8, row = (s >> 2) & 63, c8 = (s & 3) * 8;
            const bf16* w = g ? wib : wfb;
            cp_b128(&w[(size_t)(nBase + row) * KDIM + k0 + c8],
                    &lds_w[buf][g][row][c8]);
        }
    };

    stage(0, 0);
    int cur = 0;
    for (int k0 = 0; k0 < KDIM; k0 += 32) {
        wait_async_stage();                  // s_wait_asynccnt 0
        __syncthreads();                     // buffer `cur` visible to all waves
        if (k0 + 32 < KDIM) stage(k0 + 32, cur ^ 1);  // overlap next tile

        Frag16 af[2];
        #pragma unroll
        for (int r = 0; r < 2; ++r) {
            int row = wv * 32 + r * 16 + rowA;
            af[r].h[0] = *(const v8bf*)&lds_x[cur][row][kbA];
            af[r].h[1] = *(const v8bf*)&lds_x[cur][row][kbA + 16];
        }
        #pragma unroll
        for (int g = 0; g < 2; ++g)
            #pragma unroll
            for (int j = 0; j < 4; ++j) {
                Frag16 bfr;
                bfr.h[0] = *(const v8bf*)&lds_w[cur][g][j * 16 + bN][kbB];
                bfr.h[1] = *(const v8bf*)&lds_w[cur][g][j * 16 + bN][kbB + 16];
                #pragma unroll
                for (int r = 0; r < 2; ++r)
                    acc[g][r][j] = __builtin_amdgcn_wmma_f32_16x16x32_bf16(
                        false, af[r].v, false, bfr.v, (short)0,
                        acc[g][r][j], false, false);
            }
        __syncthreads();                     // reads of `cur` done before reuse
        cur ^= 1;
    }

    // in-register gate epilogue: a = clip(sigmoid(f_pre)), b = (1-f)*i_pre
    // C/D layout: VGPR v, lanes 0-15 -> M=v, N=lane; lanes 16-31 -> M=v+8
    #pragma unroll
    for (int j = 0; j < 4; ++j) {
        int col = nBase + j * 16 + (lane & 15);
        float bfv = bias_f[col], biv = bias_i[col];
        #pragma unroll
        for (int r = 0; r < 2; ++r)
            #pragma unroll
            for (int v = 0; v < 8; ++v) {
                int row = mBase + wv * 32 + r * 16 + v + ((lane >> 4) << 3);
                float fp = acc[0][r][j][v] + bfv;
                float ip = acc[1][r][j][v] + biv;
                float fg = 1.0f / (1.0f + __expf(-fp));
                float av = fminf(fmaxf(fg, EPSV), 1.0f - EPSV);
                float bv = (1.0f - fg) * ip;
                size_t o = (size_t)row * H_SZ + col;
                a_buf[o] = av;
                b_out[o] = bv;
            }
    }
}

// ---- scan pass A: per-chunk affine composition (A,B) ------------------------
__global__ __launch_bounds__(256)
void scanA_kernel(const float* __restrict__ a_buf,
                  const float* __restrict__ b_buf,
                  float* __restrict__ chunkA,
                  float* __restrict__ chunkB)
{
    int h    = (blockIdx.x & 3) * 256 + threadIdx.x;   // H = 1024 = 4*256
    int rest = blockIdx.x >> 2;
    int ch   = rest % NCHUNK;
    int b0   = rest / NCHUNK;
    size_t m0 = (size_t)b0 * S_SZ + (size_t)ch * CHLEN;
    float A = 1.0f, Bc = 0.0f;
    for (int s = 0; s < CHLEN; ++s) {
        size_t o = (m0 + s) * H_SZ + h;
        float av = a_buf[o], bv = b_buf[o];
        A  = av * A;
        Bc = av * Bc + bv;
    }
    size_t co = ((size_t)b0 * NCHUNK + ch) * H_SZ + h;
    chunkA[co] = A;
    chunkB[co] = Bc;
}

// ---- scan pass B: serial prefix over chunks per (batch, h) ------------------
__global__ __launch_bounds__(256)
void scanB_kernel(const float* __restrict__ chunkA,
                  const float* __restrict__ chunkB,
                  float* __restrict__ carryIn)
{
    int h  = (blockIdx.x & 3) * 256 + threadIdx.x;
    int b0 = blockIdx.x >> 2;
    float carry = 0.0f;   // c_{-1} = 0
    for (int ch = 0; ch < NCHUNK; ++ch) {
        size_t o = ((size_t)b0 * NCHUNK + ch) * H_SZ + h;
        carryIn[o] = carry;
        carry = chunkA[o] * carry + chunkB[o];
    }
}

// ---- scan pass C: apply recurrence in place over d_out ----------------------
__global__ __launch_bounds__(256)
void scanC_kernel(const float* __restrict__ a_buf,
                  const float* __restrict__ carryIn,
                  float* __restrict__ c_out)
{
    int h    = (blockIdx.x & 3) * 256 + threadIdx.x;
    int rest = blockIdx.x >> 2;
    int ch   = rest % NCHUNK;
    int b0   = rest / NCHUNK;
    size_t m0 = (size_t)b0 * S_SZ + (size_t)ch * CHLEN;
    float c = carryIn[((size_t)b0 * NCHUNK + ch) * H_SZ + h];
    for (int s = 0; s < CHLEN; ++s) {
        size_t o = (m0 + s) * H_SZ + h;
        c = a_buf[o] * c + c_out[o];   // c_out currently holds b_t
        c_out[o] = c;
    }
}

extern "C" void kernel_launch(void* const* d_in, const int* in_sizes, int n_in,
                              void* d_out, int out_size, void* d_ws, size_t ws_size,
                              hipStream_t stream) {
    (void)in_sizes; (void)n_in; (void)out_size; (void)ws_size;
    const float* x   = (const float*)d_in[0];   // (B,S,D)
    const float* W_f = (const float*)d_in[1];   // (H,D)
    const float* b_f = (const float*)d_in[2];   // (H)
    const float* W_i = (const float*)d_in[3];   // (H,D)
    const float* b_i = (const float*)d_in[4];   // (H)
    float* out = (float*)d_out;                 // (B,S,H)

    // workspace carve-out
    char* ws = (char*)d_ws;
    size_t off = 0;
    bf16* xb   = (bf16*)(ws + off); off += (size_t)M_SZ * D_SZ * sizeof(bf16);
    bf16* wfb  = (bf16*)(ws + off); off += (size_t)H_SZ * D_SZ * sizeof(bf16);
    bf16* wib  = (bf16*)(ws + off); off += (size_t)H_SZ * D_SZ * sizeof(bf16);
    float* a_buf   = (float*)(ws + off); off += (size_t)M_SZ * H_SZ * sizeof(float);
    float* chunkA  = (float*)(ws + off); off += (size_t)B_SZ * NCHUNK * H_SZ * sizeof(float);
    float* chunkB  = (float*)(ws + off); off += (size_t)B_SZ * NCHUNK * H_SZ * sizeof(float);
    float* carryIn = (float*)(ws + off); off += (size_t)B_SZ * NCHUNK * H_SZ * sizeof(float);

    // 1) fp32 -> bf16 conversions (vectorized, 4 elems per iter)
    cvt_bf16_kernel<<<1024, 256, 0, stream>>>(x,   xb,  (M_SZ * D_SZ) / 4);
    cvt_bf16_kernel<<<256,  256, 0, stream>>>(W_f, wfb, (H_SZ * D_SZ) / 4);
    cvt_bf16_kernel<<<256,  256, 0, stream>>>(W_i, wib, (H_SZ * D_SZ) / 4);

    // 2) fused dual GEMM + gates: a -> a_buf, b -> out
    dim3 ggrid(H_SZ / NB, M_SZ / MB);   // (16, 64)
    gemm_gates_kernel<<<ggrid, 256, 0, stream>>>(xb, wfb, wib, b_f, b_i, a_buf, out);

    // 3) chunked linear scan along S (in place over out)
    int nAC = B_SZ * NCHUNK * (H_SZ / 256);   // 512 blocks
    scanA_kernel<<<nAC, 256, 0, stream>>>(a_buf, out, chunkA, chunkB);
    scanB_kernel<<<B_SZ * (H_SZ / 256), 256, 0, stream>>>(chunkA, chunkB, carryIn);
    scanC_kernel<<<nAC, 256, 0, stream>>>(a_buf, carryIn, out);
}
